// LinearAttention_25847113187589
// MI455X (gfx1250) — matrix operand order repaired and checked by reading, not compiled, not run on hardware
//
#include <hip/hip_runtime.h>

typedef __bf16 bf16;
typedef __bf16 bf16x4 __attribute__((ext_vector_type(4)));
typedef __bf16 bf16x8 __attribute__((ext_vector_type(8)));
typedef __bf16 v16bf  __attribute__((ext_vector_type(16)));
typedef float  v8f    __attribute__((ext_vector_type(8)));
typedef float  f32x4  __attribute__((ext_vector_type(4)));

#define B_   4
#define L_   8192
#define D_   1024
#define N3_  3072

static __device__ __forceinline__ float phi_elu1(float x) {
    // elu(x)+1 : x>0 ? x+1 : exp(x)
    return x > 0.0f ? x + 1.0f : __expf(x);
}

// ---------------------------------------------------------------------------
// Conversion kernels: fp32 -> bf16, and W transposed to [3D][D] (K-contiguous)
// ---------------------------------------------------------------------------
__global__ void la_cvt_x(const float* __restrict__ x, bf16* __restrict__ xb) {
    int i = (blockIdx.x * blockDim.x + threadIdx.x) * 4;
    f32x4 v = *(const f32x4*)(x + i);
    bf16x4 o;
    o[0] = (bf16)v[0]; o[1] = (bf16)v[1]; o[2] = (bf16)v[2]; o[3] = (bf16)v[3];
    *(bf16x4*)(xb + i) = o;
}

__global__ void la_cvt_w(const float* __restrict__ W, bf16* __restrict__ Wt) {
    int idx = blockIdx.x * blockDim.x + threadIdx.x;   // over N3_*D_
    int n = idx >> 10;          // /D_
    int k = idx & (D_ - 1);     // %D_
    Wt[idx] = (bf16)W[(size_t)k * N3_ + n];            // Wt[n][k]
}

// ---------------------------------------------------------------------------
// Shared GEMM fragment helpers.
//  A-frag (16x32 bf16, row-major-in-K source): lane (m,half) needs two
//  contiguous 8-elem K-chunks at k+8*half and k+16+8*half.
//  B-frag (32x16 bf16, [N][K] source): lane (n,half) needs the same two
//  8-elem K-chunks of column n at k+8*half and k+16+8*half.
// ---------------------------------------------------------------------------
static __device__ __forceinline__ v16bf load_a_frag(const bf16* p) {
    bf16x8 lo = *(const bf16x8*)p;
    bf16x8 hi = *(const bf16x8*)(p + 16);
    return __builtin_shufflevector(lo, hi, 0,1,2,3,4,5,6,7,8,9,10,11,12,13,14,15);
}

// ---------------------------------------------------------------------------
// GEMM 1: qkv = xb @ W  (M=B*L, N=3D, K=D), fused elu+1 epilogue.
//  q  -> phiQ  [B][L][D]   row-major (A-side of GEMM 3)
//  k  -> phiKt [B][D][L]   transposed (A-side of GEMM 2)
//  v  -> Vt    [B][D][L]   transposed (B-side of GEMM 2)
// Workgroup: 8 waves (4x2), tile 256(M) x 128(N); wave tile 64x64.
// ---------------------------------------------------------------------------
__global__ __launch_bounds__(256) void la_qkv(
    const bf16* __restrict__ xb, const bf16* __restrict__ Wt,
    bf16* __restrict__ phiQ, bf16* __restrict__ phiKt, bf16* __restrict__ Vt)
{
    const int tid = threadIdx.x;
    const int wave = tid >> 5, lane = tid & 31;
    const int lane_lo = lane & 15, half = lane >> 4;
    const int wm = wave >> 1, wn = wave & 1;          // 4x2 wave grid
    const int m0 = blockIdx.x * 256 + wm * 64;        // row in B*L
    const int n0 = blockIdx.y * 128 + wn * 64;        // col in 3D

    const v8f vzero = {0.f,0.f,0.f,0.f,0.f,0.f,0.f,0.f};
    v8f acc[4][4];
#pragma unroll
    for (int i = 0; i < 4; ++i)
#pragma unroll
        for (int j = 0; j < 4; ++j) acc[i][j] = vzero;

    for (int kk = 0; kk < D_; kk += 32) {
        v16bf a[4], b[4];
#pragma unroll
        for (int i = 0; i < 4; ++i)
            a[i] = load_a_frag(xb + (size_t)(m0 + i*16 + lane_lo) * D_ + kk + 8*half);
#pragma unroll
        for (int j = 0; j < 4; ++j)
            b[j] = load_a_frag(Wt + (size_t)(n0 + j*16 + lane_lo) * D_ + kk + 8*half);
#pragma unroll
        for (int i = 0; i < 4; ++i)
#pragma unroll
            for (int j = 0; j < 4; ++j)
                acc[i][j] = __builtin_amdgcn_wmma_f32_16x16x32_bf16(
                    false, a[i], false, b[j], (short)0, acc[i][j], false, false);
    }

    // Epilogue: D-matrix layout -> lane holds column n, rows m = mt..mt+7.
#pragma unroll
    for (int i = 0; i < 4; ++i) {
        const int mt = m0 + i*16 + 8*half;             // first of 8 rows
        const int bb = mt / L_;                        // batch (uniform over r)
        const int l  = mt % L_;
#pragma unroll
        for (int j = 0; j < 4; ++j) {
            const int col = n0 + j*16 + lane_lo;
            if (col < D_) {                            // ---- q: phi + row-major
                const int d = col;
#pragma unroll
                for (int r = 0; r < 8; ++r) {
                    float v = phi_elu1(acc[i][j][r]);
                    phiQ[((size_t)bb * L_ + (l + r)) * D_ + d] = (bf16)v;
                }
            } else if (col < 2*D_) {                   // ---- k: phi + transposed
                const int d = col - D_;
                bf16x8 o;
#pragma unroll
                for (int r = 0; r < 8; ++r) o[r] = (bf16)phi_elu1(acc[i][j][r]);
                *(bf16x8*)(phiKt + ((size_t)bb * D_ + d) * L_ + l) = o;
            } else {                                   // ---- v: transposed
                const int h = col - 2*D_;
                bf16x8 o;
#pragma unroll
                for (int r = 0; r < 8; ++r) o[r] = (bf16)acc[i][j][r];
                *(bf16x8*)(Vt + ((size_t)bb * D_ + h) * L_ + l) = o;
            }
        }
    }
}

// ---------------------------------------------------------------------------
// GEMM 2: KV[b][h][d] = sum_l phiK[l][d] * v[l][h]   (M=d, N=h, K=L)
//   A = phiKt [d][l] (K-contig), B = Vt [h][l] (K-contig), out bf16 [h][d].
// ---------------------------------------------------------------------------
__global__ __launch_bounds__(256) void la_kv(
    const bf16* __restrict__ phiKt, const bf16* __restrict__ Vt,
    bf16* __restrict__ KVb)
{
    const int tid = threadIdx.x;
    const int wave = tid >> 5, lane = tid & 31;
    const int lane_lo = lane & 15, half = lane >> 4;
    const int wm = wave >> 1, wn = wave & 1;
    const int bz = blockIdx.z;
    const int m0 = blockIdx.x * 256 + wm * 64;   // d
    const int n0 = blockIdx.y * 128 + wn * 64;   // h
    const bf16* Ab = phiKt + (size_t)bz * D_ * L_;
    const bf16* Bb = Vt    + (size_t)bz * D_ * L_;

    const v8f vzero = {0.f,0.f,0.f,0.f,0.f,0.f,0.f,0.f};
    v8f acc[4][4];
#pragma unroll
    for (int i = 0; i < 4; ++i)
#pragma unroll
        for (int j = 0; j < 4; ++j) acc[i][j] = vzero;

    for (int kk = 0; kk < L_; kk += 32) {
        v16bf a[4], b[4];
#pragma unroll
        for (int i = 0; i < 4; ++i)
            a[i] = load_a_frag(Ab + (size_t)(m0 + i*16 + lane_lo) * L_ + kk + 8*half);
#pragma unroll
        for (int j = 0; j < 4; ++j)
            b[j] = load_a_frag(Bb + (size_t)(n0 + j*16 + lane_lo) * L_ + kk + 8*half);
#pragma unroll
        for (int i = 0; i < 4; ++i)
#pragma unroll
            for (int j = 0; j < 4; ++j)
                acc[i][j] = __builtin_amdgcn_wmma_f32_16x16x32_bf16(
                    false, a[i], false, b[j], (short)0, acc[i][j], false, false);
    }

    // store KV[b][h][d] bf16: lane fixed h = n, d = dt..dt+7 contiguous
#pragma unroll
    for (int i = 0; i < 4; ++i) {
        const int dt = m0 + i*16 + 8*half;
#pragma unroll
        for (int j = 0; j < 4; ++j) {
            const int h = n0 + j*16 + lane_lo;
            bf16x8 o;
#pragma unroll
            for (int r = 0; r < 8; ++r) o[r] = (bf16)acc[i][j][r];
            *(bf16x8*)(KVb + ((size_t)bz * D_ + h) * D_ + dt) = o;
        }
    }
}

// ---------------------------------------------------------------------------
// k_sum[b][d] = sum_l phiK (read from phiKt rows, contiguous)
// ---------------------------------------------------------------------------
__global__ void la_ksum(const bf16* __restrict__ phiKt, float* __restrict__ ksum) {
    __shared__ float red[256];
    const int row = blockIdx.x;                     // b*D + d
    const bf16* p = phiKt + (size_t)row * L_;
    float s = 0.f;
    for (int l = threadIdx.x; l < L_; l += 256) s += (float)p[l];
    red[threadIdx.x] = s;
    __syncthreads();
    for (int off = 128; off > 0; off >>= 1) {
        if ((int)threadIdx.x < off) red[threadIdx.x] += red[threadIdx.x + off];
        __syncthreads();
    }
    if (threadIdx.x == 0) ksum[row] = red[0];
}

// ---------------------------------------------------------------------------
// Z[b][l] = 1 / (phiQ[b][l][:] . ksum[b][:] + eps)   — one wave32 per row
// ---------------------------------------------------------------------------
__global__ void la_z(const bf16* __restrict__ phiQ, const float* __restrict__ ksum,
                     float* __restrict__ Z) {
    const int row  = blockIdx.x * 8 + (threadIdx.x >> 5);  // b*L + l
    const int lane = threadIdx.x & 31;
    const int b = row / L_;
    const bf16*  q  = phiQ + (size_t)row * D_;
    const float* ks = ksum + b * D_;
    float s = 0.f;
    for (int d = lane; d < D_; d += 32) s += (float)q[d] * ks[d];
#pragma unroll
    for (int off = 16; off > 0; off >>= 1) s += __shfl_xor(s, off, 32);
    if (lane == 0) Z[row] = 1.0f / (s + 1e-6f);
}

// ---------------------------------------------------------------------------
// GEMM 3: out[b][l][h] = Z[b][l] * sum_d phiQ[l][d] * KV[h][d]
//   A = phiQ [l][d] (K-contig), B = KVb [h][d] (K-contig), fp32 epilogue * Z.
// ---------------------------------------------------------------------------
__global__ __launch_bounds__(256) void la_out(
    const bf16* __restrict__ phiQ, const bf16* __restrict__ KVb,
    const float* __restrict__ Z, float* __restrict__ out)
{
    const int tid = threadIdx.x;
    const int wave = tid >> 5, lane = tid & 31;
    const int lane_lo = lane & 15, half = lane >> 4;
    const int wm = wave >> 1, wn = wave & 1;
    const int bz = blockIdx.z;
    const int m0 = blockIdx.x * 256 + wm * 64;   // l
    const int n0 = blockIdx.y * 128 + wn * 64;   // h
    const bf16* Ab = phiQ + (size_t)bz * L_ * D_;
    const bf16* Bb = KVb  + (size_t)bz * D_ * D_;

    const v8f vzero = {0.f,0.f,0.f,0.f,0.f,0.f,0.f,0.f};
    v8f acc[4][4];
#pragma unroll
    for (int i = 0; i < 4; ++i)
#pragma unroll
        for (int j = 0; j < 4; ++j) acc[i][j] = vzero;

    for (int kk = 0; kk < D_; kk += 32) {
        v16bf a[4], b[4];
#pragma unroll
        for (int i = 0; i < 4; ++i)
            a[i] = load_a_frag(Ab + (size_t)(m0 + i*16 + lane_lo) * D_ + kk + 8*half);
#pragma unroll
        for (int j = 0; j < 4; ++j)
            b[j] = load_a_frag(Bb + (size_t)(n0 + j*16 + lane_lo) * D_ + kk + 8*half);
#pragma unroll
        for (int i = 0; i < 4; ++i)
#pragma unroll
            for (int j = 0; j < 4; ++j)
                acc[i][j] = __builtin_amdgcn_wmma_f32_16x16x32_bf16(
                    false, a[i], false, b[j], (short)0, acc[i][j], false, false);
    }

    // epilogue: per VGPR r, lanes 0..15 write 16 consecutive h -> coalesced
#pragma unroll
    for (int i = 0; i < 4; ++i) {
        const int lt = m0 + i*16 + 8*half;
#pragma unroll
        for (int j = 0; j < 4; ++j) {
            const int h = n0 + j*16 + lane_lo;
#pragma unroll
            for (int r = 0; r < 8; ++r) {
                const int l = lt + r;
                const float z = Z[bz * L_ + l];
                out[((size_t)bz * L_ + l) * D_ + h] = acc[i][j][r] * z;
            }
        }
    }
}

// ---------------------------------------------------------------------------
extern "C" void kernel_launch(void* const* d_in, const int* in_sizes, int n_in,
                              void* d_out, int out_size, void* d_ws, size_t ws_size,
                              hipStream_t stream) {
    const float* x = (const float*)d_in[0];   // [B, L, D]
    const float* W = (const float*)d_in[1];   // [D, 3D]
    float* out = (float*)d_out;               // [B, L, D]

    char* ws = (char*)d_ws;
    size_t off = 0;
    bf16* xb    = (bf16*)(ws + off); off += (size_t)B_ * L_ * D_ * 2;   // 64 MB
    bf16* Wt    = (bf16*)(ws + off); off += (size_t)N3_ * D_ * 2;       //  6 MB
    bf16* phiQ  = (bf16*)(ws + off); off += (size_t)B_ * L_ * D_ * 2;   // 64 MB
    bf16* phiKt = (bf16*)(ws + off); off += (size_t)B_ * L_ * D_ * 2;   // 64 MB
    bf16* Vt    = (bf16*)(ws + off); off += (size_t)B_ * L_ * D_ * 2;   // 64 MB
    bf16* KVb   = (bf16*)(ws + off); off += (size_t)B_ * D_ * D_ * 2;   //  8 MB
    float* ksum = (float*)(ws + off); off += (size_t)B_ * D_ * 4;
    float* Zv   = (float*)(ws + off); off += (size_t)B_ * L_ * 4;

    // 1) fp32 -> bf16 conversions
    la_cvt_x<<<(B_ * L_ * D_) / (256 * 4), 256, 0, stream>>>(x, xb);
    la_cvt_w<<<(N3_ * D_) / 256, 256, 0, stream>>>(W, Wt);

    // 2) qkv GEMM + fused phi + layout-transforming epilogue
    la_qkv<<<dim3((B_ * L_) / 256, N3_ / 128), 256, 0, stream>>>(xb, Wt, phiQ, phiKt, Vt);

    // 3) KV = phiK^T @ v  per batch
    la_kv<<<dim3(D_ / 256, D_ / 128, B_), 256, 0, stream>>>(phiKt, Vt, KVb);

    // 4) k_sum and Z
    la_ksum<<<B_ * D_, 256, 0, stream>>>(phiKt, ksum);
    la_z<<<(B_ * L_) / 8, 256, 0, stream>>>(phiQ, ksum, Zv);

    // 5) out = (phiQ @ KV^T) * Z
    la_out<<<dim3(L_ / 256, D_ / 128, B_), 256, 0, stream>>>(phiQ, KVb, Zv, out);
}
